// Mol2SpecEGNN_85117661872938
// MI455X (gfx1250) — matrix operand order repaired and checked by reading, not compiled, not run on hardware
//
#include <hip/hip_runtime.h>
#include <math.h>

#define N_NODES 20000
#define N_EDGES 160000
#define N_BATCH 1024
#define XDIM    64
#define EDIM    16
#define HDIM    128
#define NLAYER  4
#define PROPN   1000

typedef __attribute__((ext_vector_type(16))) _Float16 v16h;
typedef __attribute__((ext_vector_type(8)))  _Float16 v8h;
typedef __attribute__((ext_vector_type(4)))  _Float16 v4h;
typedef __attribute__((ext_vector_type(8)))  float    v8f;
typedef __attribute__((ext_vector_type(4)))  int      v4i;

// ------------------------------------------------------------ async G->LDS path
#if defined(__HIP_DEVICE_COMPILE__) && __has_builtin(__builtin_amdgcn_global_load_async_to_lds_b128)
#define HAVE_ASYNC_LDS 1
typedef __attribute__((address_space(1))) v4i as1_v4i;
typedef __attribute__((address_space(3))) v4i as3_v4i;
#else
#define HAVE_ASYNC_LDS 0
#endif

__device__ __forceinline__ void wait_async0() {
#if defined(__HIP_DEVICE_COMPILE__)
#if __has_builtin(__builtin_amdgcn_s_wait_asynccnt)
  __builtin_amdgcn_s_wait_asynccnt(0);
#else
  asm volatile("s_wait_asynccnt 0x0" ::: "memory");
#endif
#endif
}

// Workgroup-cooperative copy of f16 weights into LDS (b128 granules).
__device__ __forceinline__ void wg_copy_b128(_Float16* __restrict__ dst,
                                             const _Float16* __restrict__ src,
                                             int halves, int tid) {
  const int vecs = halves >> 3;
#if HAVE_ASYNC_LDS
  for (int i = tid; i < vecs; i += 256)
    __builtin_amdgcn_global_load_async_to_lds_b128(
        (as1_v4i*)(src + (size_t)i * 8), (as3_v4i*)(dst + (size_t)i * 8), 0, 0);
#else
  for (int i = tid; i < vecs; i += 256)
    ((uint4*)dst)[i] = ((const uint4*)src)[i];
#endif
}

__device__ __forceinline__ void wg_copy_done() {
#if HAVE_ASYNC_LDS
  wait_async0();
#endif
  __syncthreads();
}

// ---------------------------------------------------------------- utilities
__device__ __forceinline__ float silu_f(float x) { return x / (1.f + __expf(-x)); }

__device__ __forceinline__ v8f v8f_zero() {
  v8f z;
#pragma unroll
  for (int i = 0; i < 8; ++i) z[i] = 0.f;
  return z;
}

__device__ __forceinline__ v4h cvt4(float4 f) {
  v4h r;
  r[0] = (_Float16)f.x; r[1] = (_Float16)f.y;
  r[2] = (_Float16)f.z; r[3] = (_Float16)f.w;
  return r;
}

// Order LDS traffic within a wave (cross-lane reuse of wave-private LDS tiles).
__device__ __forceinline__ void wave_sync() {
  __builtin_amdgcn_fence(__ATOMIC_ACQ_REL, "wavefront");
  __builtin_amdgcn_wave_barrier();
}

__device__ __forceinline__ void atomicMaxF(float* addr, float val) {
  unsigned int* ua = (unsigned int*)addr;
  unsigned int old = *ua;
  while (true) {
    float f = __uint_as_float(old);
    if (f >= val) break;
    unsigned int assumed = old;
    old = atomicCAS(ua, assumed, __float_as_uint(val));
    if (old == assumed) break;
  }
}

// A fragment (16x32 f16): lane = kg*16+m holds k = kc*32 + kg*8 + {0..7, 16..23}
__device__ __forceinline__ v16h load_afrag(const _Float16* A, int lda, int kc, int lane) {
  const int m = lane & 15, kg = lane >> 4;
  const _Float16* ab = A + m * lda + kc * 32 + kg * 8;
  v8h a0 = *(const v8h*)(ab);
  v8h a1 = *(const v8h*)(ab + 16);
  v16h a;
#pragma unroll
  for (int i = 0; i < 8; ++i) { a[i] = a0[i]; a[8 + i] = a1[i]; }
  return a;
}

// One wave computes a 16 x (NT*16) GEMM tile:  C += A(16 x KT*32) * B.
// A: LDS, row-major halves with stride LDA.  B: packed fragment tiles, 512 halves each.
template<int KT, int NT, int LDA>
__device__ __forceinline__ void wave_gemm(const _Float16* __restrict__ A,
                                          const _Float16* __restrict__ B,
                                          v8f acc[NT], int lane) {
#pragma unroll
  for (int kc = 0; kc < KT; ++kc) {
    v16h a = load_afrag(A, LDA, kc, lane);
#pragma unroll
    for (int nt = 0; nt < NT; ++nt) {
      v16h b = *(const v16h*)(B + (kc * NT + nt) * 512 + lane * 16);
      acc[nt] = __builtin_amdgcn_wmma_f32_16x16x32_f16(
          false, a, false, b, (short)0, acc[nt], false, false);
    }
  }
}

// C frags (f32) -> bias -> SiLU -> f16 tile in LDS (A-layout for next GEMM).
template<int NT>
__device__ __forceinline__ void emit_silu_f16(const v8f acc[NT], const float* __restrict__ bias,
                                              _Float16* __restrict__ T, int ldt, int lane) {
  const int n0 = lane & 15, mg = (lane >> 4) * 8;
#pragma unroll
  for (int nt = 0; nt < NT; ++nt) {
    const int n = nt * 16 + n0;
    const float bv = bias[n];
#pragma unroll
    for (int r = 0; r < 8; ++r)
      T[(mg + r) * ldt + n] = (_Float16)silu_f(acc[nt][r] + bv);
  }
}

// ---------------------------------------------------------------- small kernels
__global__ void fill_kernel(float* p, float v, int n) {
  int i = blockIdx.x * blockDim.x + threadIdx.x;
  if (i < n) p[i] = v;
}

__global__ void count_kernel(const int* __restrict__ row, float* __restrict__ cnt, int nE) {
  int i = blockIdx.x * blockDim.x + threadIdx.x;
  if (i < nE) atomicAdd(&cnt[row[i]], 1.f);
}

__global__ void pos_update_kernel(float* __restrict__ pos, float* __restrict__ posacc,
                                  const float* __restrict__ cnt, int nN) {
  int i = blockIdx.x * blockDim.x + threadIdx.x;
  if (i >= nN * 3) return;
  float c = fmaxf(cnt[i / 3], 1.f);
  pos[i] += posacc[i] / c;
  posacc[i] = 0.f;
}

// Convert f32 weight [K x Nfull] (row-major, x@W convention) into f16 WMMA B-fragment
// tiles: packed[(kc*NT+nt)*512 + lane*16 + i] = W[kc*32 + (i>>3)*16 + (lane>>4)*8 + (i&7)][nt*16 + (lane&15)]
__global__ void pack_w_kernel(const float* __restrict__ W, _Float16* __restrict__ P,
                              int K, int Nfull, int KT, int NT) {
  int idx = blockIdx.x * blockDim.x + threadIdx.x;
  int total = KT * NT * 512;
  if (idx >= total) return;
  int tile = idx >> 9, pos = idx & 511;
  int lane = pos >> 4, i = pos & 15;
  int kc = tile / NT, nt = tile - kc * NT;
  int n = nt * 16 + (lane & 15);
  int k = kc * 32 + ((i >> 3) << 4) + ((lane >> 4) << 3) + (i & 7);
  float v = (k < K && n < Nfull) ? W[(size_t)k * Nfull + n] : 0.f;
  P[idx] = (_Float16)v;
}

// ---------------------------------------------------------------- emb_in: h = x@W + b
__launch_bounds__(256)
__global__ void emb_in_kernel(const float* __restrict__ x, const _Float16* __restrict__ wp,
                              const float* __restrict__ b, float* __restrict__ h, int nN) {
  extern __shared__ __align__(128) char smem[];
  _Float16* W   = (_Float16*)smem;     // 16 tiles * 512
  _Float16* IN0 = W + 16 * 512;        // per-wave 16*64
  const int tid = threadIdx.x, lane = tid & 31, wid = tid >> 5;
  wg_copy_b128(W, wp, 16 * 512, tid);
  wg_copy_done();
  _Float16* IN = IN0 + wid * (16 * 64);
  const int ntiles = (nN + 15) >> 4;
  const int gw = blockIdx.x * 8 + wid, nw = gridDim.x * 8;
  const int nc = lane & 15, mg = (lane >> 4) * 8;
  const float4 fz = {0.f, 0.f, 0.f, 0.f};
  for (int tile = gw; tile < ntiles; tile += nw) {
    const int n0 = tile << 4;
    for (int t = lane; t < 16 * 16; t += 32) {     // 16 rows x 16 float4
      const int m = t >> 4, q = t & 15;
      const int node = n0 + m;
      const float4 xv = (node < nN) ? ((const float4*)(x + (size_t)node * XDIM))[q] : fz;
      *(v4h*)(IN + m * 64 + q * 4) = cvt4(xv);
    }
    wave_sync();
    v8f a1[8];
#pragma unroll
    for (int i = 0; i < 8; ++i) a1[i] = v8f_zero();
    wave_gemm<2, 8, 64>(IN, W, a1, lane);
#pragma unroll
    for (int nt = 0; nt < 8; ++nt) {
      const int n = nt * 16 + nc;
      const float bv = b[n];
#pragma unroll
      for (int r = 0; r < 8; ++r) {
        const int node = n0 + mg + r;
        if (node < nN) h[(size_t)node * HDIM + n] = a1[nt][r] + bv;
      }
    }
    wave_sync();
  }
}

// ---------------------------------------------------------------- edge model (fused)
// per edge: e_in=[h[row],h[col],radial,eattr] -> silu(silu(.)W2) = m ; agg[row]+=m
//           p = silu(m Wc1) ; s = p.wc2 ; posacc[row] += coord_diff * s
__launch_bounds__(256)
__global__ void edge_kernel(const float* __restrict__ h, const float* __restrict__ pos,
                            const float* __restrict__ eattr,
                            const int* __restrict__ row, const int* __restrict__ col,
                            const _Float16* __restrict__ w1p, const float* __restrict__ b1,
                            const _Float16* __restrict__ w2p, const float* __restrict__ b2,
                            const _Float16* __restrict__ wc1p, const float* __restrict__ bc1,
                            const float* __restrict__ wc2,
                            float* __restrict__ agg, float* __restrict__ posacc, int nE) {
  extern __shared__ __align__(128) char smem[];
  _Float16* W1   = (_Float16*)smem;      // 72*512
  _Float16* W2   = W1 + 72 * 512;        // 32*512
  _Float16* WC   = W2 + 32 * 512;        // 32*512
  _Float16* EIN0 = WC + 32 * 512;        // per-wave 16*288
  _Float16* T0   = EIN0 + 8 * 4608;      // per-wave 16*128
  const int tid = threadIdx.x, lane = tid & 31, wid = tid >> 5;
  wg_copy_b128(W1, w1p, 72 * 512, tid);
  wg_copy_b128(W2, w2p, 32 * 512, tid);
  wg_copy_b128(WC, wc1p, 32 * 512, tid);
  wg_copy_done();
  _Float16* EIN = EIN0 + wid * 4608;
  _Float16* T   = T0 + wid * 2048;
  const int ntiles = nE >> 4;
  const int gw = blockIdx.x * 8 + wid, nw = gridDim.x * 8;
  for (int tile = gw; tile < ntiles; tile += nw) {
    const int e0 = tile << 4;
    if (tile + nw < ntiles) __builtin_prefetch(&row[(tile + nw) << 4], 0, 0);
    // stage e_in [16 x 288] f16: vectorized b128 gathers of h rows
#pragma unroll 1
    for (int m = 0; m < 16; ++m) {
      const int e = e0 + m;
      const float4 hr = ((const float4*)(h + (size_t)row[e] * HDIM))[lane];
      const float4 hc = ((const float4*)(h + (size_t)col[e] * HDIM))[lane];
      *(v4h*)(EIN + m * 288 + lane * 4)       = cvt4(hr);
      *(v4h*)(EIN + m * 288 + 128 + lane * 4) = cvt4(hc);
    }
    for (int t = lane; t < 16 * EDIM; t += 32) {
      const int m = t >> 4, k = t & 15;
      EIN[m * 288 + 257 + k] = (_Float16)eattr[(size_t)(e0 + m) * EDIM + k];
    }
    if (lane < 16) {
      const int e = e0 + lane;
      const int ir = row[e], ic = col[e];
      const float dx = pos[ir * 3 + 0] - pos[ic * 3 + 0];
      const float dy = pos[ir * 3 + 1] - pos[ic * 3 + 1];
      const float dz = pos[ir * 3 + 2] - pos[ic * 3 + 2];
      EIN[lane * 288 + 256] = (_Float16)(dx * dx + dy * dy + dz * dz);
#pragma unroll
      for (int k = 273; k < 288; ++k) EIN[lane * 288 + k] = (_Float16)0.f;
    }
    wave_sync();
    v8f a1[8];
#pragma unroll
    for (int i = 0; i < 8; ++i) a1[i] = v8f_zero();
    wave_gemm<9, 8, 288>(EIN, W1, a1, lane);          // [16,288]@[288,128]
    emit_silu_f16<8>(a1, b1, T, 128, lane);
    wave_sync();
    v8f a2[8];
#pragma unroll
    for (int i = 0; i < 8; ++i) a2[i] = v8f_zero();
    wave_gemm<4, 8, 128>(T, W2, a2, lane);            // [16,128]@[128,128]
    emit_silu_f16<8>(a2, b2, T, 128, lane);           // T = m
    wave_sync();
    v8f a3[8];
#pragma unroll
    for (int i = 0; i < 8; ++i) a3[i] = v8f_zero();
    wave_gemm<4, 8, 128>(T, WC, a3, lane);            // m @ coord_w1
    // scatter m into agg[row] while m is still in T
    for (int t = lane; t < 16 * 128; t += 32) {
      const int m = t >> 7, n = t & 127;
      atomicAdd(&agg[(size_t)row[e0 + m] * HDIM + n], (float)T[t]);
    }
    wave_sync();
    emit_silu_f16<8>(a3, bc1, T, 128, lane);          // T = p
    wave_sync();
    if (lane < 16) {
      const int e = e0 + lane;
      float s = 0.f;
      for (int n = 0; n < 128; ++n) s += (float)T[lane * 128 + n] * wc2[n];
      const int ir = row[e], ic = col[e];
      const float dx = pos[ir * 3 + 0] - pos[ic * 3 + 0];
      const float dy = pos[ir * 3 + 1] - pos[ic * 3 + 1];
      const float dz = pos[ir * 3 + 2] - pos[ic * 3 + 2];
      atomicAdd(&posacc[ir * 3 + 0], dx * s);
      atomicAdd(&posacc[ir * 3 + 1], dy * s);
      atomicAdd(&posacc[ir * 3 + 2], dz * s);
    }
    wave_sync();
  }
}

// ---------------------------------------------------------------- node model
// h = h + silu([h,agg]@W1+b1)@W2 + b2 ; agg cleared for next layer
__launch_bounds__(256)
__global__ void node_kernel(float* __restrict__ h, float* __restrict__ agg,
                            const _Float16* __restrict__ w1p, const float* __restrict__ b1,
                            const _Float16* __restrict__ w2p, const float* __restrict__ b2,
                            int nN) {
  extern __shared__ __align__(128) char smem[];
  _Float16* W1  = (_Float16*)smem;    // 64*512
  _Float16* W2  = W1 + 64 * 512;      // 32*512
  _Float16* IN0 = W2 + 32 * 512;      // per-wave 16*256
  _Float16* T0  = IN0 + 8 * 4096;     // per-wave 16*128
  const int tid = threadIdx.x, lane = tid & 31, wid = tid >> 5;
  wg_copy_b128(W1, w1p, 64 * 512, tid);
  wg_copy_b128(W2, w2p, 32 * 512, tid);
  wg_copy_done();
  _Float16* IN = IN0 + wid * 4096;
  _Float16* T  = T0 + wid * 2048;
  const int ntiles = (nN + 15) >> 4;
  const int gw = blockIdx.x * 8 + wid, nw = gridDim.x * 8;
  const int nc = lane & 15, mg = (lane >> 4) * 8;
  const float4 fz = {0.f, 0.f, 0.f, 0.f};
  for (int tile = gw; tile < ntiles; tile += nw) {
    const int n0 = tile << 4;
#pragma unroll 1
    for (int m = 0; m < 16; ++m) {
      const int node = n0 + m;
      const bool ok = node < nN;
      const float4 hv = ok ? ((const float4*)(h + (size_t)node * HDIM))[lane] : fz;
      const float4 av = ok ? ((const float4*)(agg + (size_t)node * HDIM))[lane] : fz;
      *(v4h*)(IN + m * 256 + lane * 4)       = cvt4(hv);
      *(v4h*)(IN + m * 256 + 128 + lane * 4) = cvt4(av);
    }
    wave_sync();
    v8f a1[8];
#pragma unroll
    for (int i = 0; i < 8; ++i) a1[i] = v8f_zero();
    wave_gemm<8, 8, 256>(IN, W1, a1, lane);
    emit_silu_f16<8>(a1, b1, T, 128, lane);
    wave_sync();
    v8f a2[8];
#pragma unroll
    for (int i = 0; i < 8; ++i) a2[i] = v8f_zero();
    wave_gemm<4, 8, 128>(T, W2, a2, lane);
#pragma unroll
    for (int nt = 0; nt < 8; ++nt) {
      const int n = nt * 16 + nc;
      const float bv = b2[n];
#pragma unroll
      for (int r = 0; r < 8; ++r) {
        const int node = n0 + mg + r;
        if (node < nN) h[(size_t)node * HDIM + n] += a2[nt][r] + bv;
      }
    }
    // clear agg for the next layer
    for (int t = lane; t < 16 * 128; t += 32) {
      const int node = n0 + (t >> 7);
      if (node < nN) agg[(size_t)node * HDIM + (t & 127)] = 0.f;
    }
    wave_sync();
  }
}

// ---------------------------------------------------------------- emb_out + global max pool
__launch_bounds__(256)
__global__ void emb_out_pool_kernel(const float* __restrict__ h, const _Float16* __restrict__ wp,
                                    const float* __restrict__ b, const int* __restrict__ batch,
                                    float* __restrict__ g, int nN) {
  extern __shared__ __align__(128) char smem[];
  _Float16* W   = (_Float16*)smem;  // 32*512
  _Float16* IN0 = W + 32 * 512;     // per-wave 16*128
  const int tid = threadIdx.x, lane = tid & 31, wid = tid >> 5;
  wg_copy_b128(W, wp, 32 * 512, tid);
  wg_copy_done();
  _Float16* IN = IN0 + wid * 2048;
  const int ntiles = (nN + 15) >> 4;
  const int gw = blockIdx.x * 8 + wid, nw = gridDim.x * 8;
  const int nc = lane & 15, mg = (lane >> 4) * 8;
  const float4 fz = {0.f, 0.f, 0.f, 0.f};
  for (int tile = gw; tile < ntiles; tile += nw) {
    const int n0 = tile << 4;
#pragma unroll 1
    for (int m = 0; m < 16; ++m) {
      const int node = n0 + m;
      const float4 hv = (node < nN) ? ((const float4*)(h + (size_t)node * HDIM))[lane] : fz;
      *(v4h*)(IN + m * 128 + lane * 4) = cvt4(hv);
    }
    wave_sync();
    v8f a1[8];
#pragma unroll
    for (int i = 0; i < 8; ++i) a1[i] = v8f_zero();
    wave_gemm<4, 8, 128>(IN, W, a1, lane);
#pragma unroll
    for (int nt = 0; nt < 8; ++nt) {
      const int n = nt * 16 + nc;
      const float bv = b[n];
#pragma unroll
      for (int r = 0; r < 8; ++r) {
        const int node = n0 + mg + r;
        if (node < nN)
          atomicMaxF(&g[(size_t)batch[node] * HDIM + n], a1[nt][r] + bv);
      }
    }
    wave_sync();
  }
}

// ---------------------------------------------------------------- resblocks + head
__launch_bounds__(256)
__global__ void head_kernel(const float* __restrict__ g,
                            const _Float16* __restrict__ rw1p, const float* __restrict__ rb1,
                            const _Float16* __restrict__ rw2p, const float* __restrict__ rb2,
                            const _Float16* __restrict__ hwp, const float* __restrict__ hb,
                            float* __restrict__ out) {
  extern __shared__ __align__(128) char smem[];
  _Float16* RW1 = (_Float16*)smem;      // 2*32*512
  _Float16* RW2 = RW1 + 2 * 32 * 512;   // 2*32*512
  _Float16* G0  = RW2 + 2 * 32 * 512;   // per-wave 16*128
  _Float16* T0  = G0 + 8 * 2048;        // per-wave 16*128
  const int tid = threadIdx.x, lane = tid & 31, wid = tid >> 5;
  wg_copy_b128(RW1, rw1p, 2 * 32 * 512, tid);
  wg_copy_b128(RW2, rw2p, 2 * 32 * 512, tid);
  wg_copy_done();
  _Float16* G = G0 + wid * 2048;
  _Float16* T = T0 + wid * 2048;
  const int nc = lane & 15, mg = (lane >> 4) * 8;
  const int tile = blockIdx.x * 8 + wid;          // 64 tiles of 16 rows
  if (tile >= (N_BATCH >> 4)) return;
  const int r0 = tile << 4;
#pragma unroll 1
  for (int m = 0; m < 16; ++m)
    *(v4h*)(G + m * 128 + lane * 4) =
        cvt4(((const float4*)(g + (size_t)(r0 + m) * HDIM))[lane]);
  wave_sync();
  for (int rb = 0; rb < 2; ++rb) {
    v8f a1[8];
#pragma unroll
    for (int i = 0; i < 8; ++i) a1[i] = v8f_zero();
    wave_gemm<4, 8, 128>(G, RW1 + rb * 32 * 512, a1, lane);
    emit_silu_f16<8>(a1, rb1 + rb * HDIM, T, 128, lane);
    wave_sync();
    v8f a2[8];
#pragma unroll
    for (int i = 0; i < 8; ++i) a2[i] = v8f_zero();
    wave_gemm<4, 8, 128>(T, RW2 + rb * 32 * 512, a2, lane);
#pragma unroll
    for (int nt = 0; nt < 8; ++nt) {
      const int n = nt * 16 + nc;
      const float bv = rb2[rb * HDIM + n];
#pragma unroll
      for (int r = 0; r < 8; ++r) {
        const int idx = (mg + r) * 128 + n;
        G[idx] = (_Float16)((float)G[idx] + a2[nt][r] + bv);
      }
    }
    wave_sync();
  }
  // head: [16,128] @ [128,1000], B-frags streamed from global (L2-resident)
  v16h afr[4];
#pragma unroll
  for (int kc = 0; kc < 4; ++kc) afr[kc] = load_afrag(G, 128, kc, lane);
  for (int nt = 0; nt < 63; ++nt) {
    v8f acc = v8f_zero();
#pragma unroll
    for (int kc = 0; kc < 4; ++kc) {
      v16h bfr = *(const v16h*)(hwp + ((size_t)(kc * 63 + nt)) * 512 + lane * 16);
      acc = __builtin_amdgcn_wmma_f32_16x16x32_f16(false, afr[kc], false, bfr, (short)0,
                                                   acc, false, false);
    }
    const int n = nt * 16 + nc;
    if (n < PROPN) {
      const float bv = hb[n];
#pragma unroll
      for (int r = 0; r < 8; ++r)
        out[(size_t)(r0 + mg + r) * PROPN + n] = acc[r] + bv;
    }
  }
}

// ---------------------------------------------------------------- host
extern "C" void kernel_launch(void* const* d_in, const int* in_sizes, int n_in,
                              void* d_out, int out_size, void* d_ws, size_t ws_size,
                              hipStream_t stream) {
  (void)in_sizes; (void)n_in; (void)out_size; (void)ws_size;
  const float* x         = (const float*)d_in[0];
  const float* pos_in    = (const float*)d_in[1];
  const float* edge_attr = (const float*)d_in[2];
  const int*   eidx      = (const int*)d_in[3];
  const int*   row       = eidx;
  const int*   col       = eidx + N_EDGES;
  const int*   batch     = (const int*)d_in[4];
  const float* emb_in_w  = (const float*)d_in[5];
  const float* emb_in_b  = (const float*)d_in[6];
  const float* emb_out_w = (const float*)d_in[7];
  const float* emb_out_b = (const float*)d_in[8];
  const float* edge_w1   = (const float*)d_in[9];
  const float* edge_b1   = (const float*)d_in[10];
  const float* edge_w2   = (const float*)d_in[11];
  const float* edge_b2   = (const float*)d_in[12];
  const float* node_w1   = (const float*)d_in[13];
  const float* node_b1   = (const float*)d_in[14];
  const float* node_w2   = (const float*)d_in[15];
  const float* node_b2   = (const float*)d_in[16];
  const float* coord_w1  = (const float*)d_in[17];
  const float* coord_b1  = (const float*)d_in[18];
  const float* coord_w2  = (const float*)d_in[19];
  const float* res_w1    = (const float*)d_in[20];
  const float* res_b1    = (const float*)d_in[21];
  const float* res_w2    = (const float*)d_in[22];
  const float* res_b2    = (const float*)d_in[23];
  const float* head_w    = (const float*)d_in[24];
  const float* head_b    = (const float*)d_in[25];
  float* out = (float*)d_out;

  // workspace carve-out (256B aligned)
  size_t off = 0;
  char* base = (char*)d_ws;
  auto alloc = [&](size_t bytes) {
    void* p = base + off;
    off += bytes;
    off = (off + 255) & ~(size_t)255;
    return p;
  };
  float* h      = (float*)alloc((size_t)N_NODES * HDIM * 4);
  float* agg    = (float*)alloc((size_t)N_NODES * HDIM * 4);
  float* posw   = (float*)alloc((size_t)N_NODES * 3 * 4);
  float* posacc = (float*)alloc((size_t)N_NODES * 3 * 4);
  float* cnt    = (float*)alloc((size_t)N_NODES * 4);
  float* g      = (float*)alloc((size_t)N_BATCH * HDIM * 4);
  _Float16* p_embin  = (_Float16*)alloc((size_t)2 * 8 * 512 * 2);
  _Float16* p_embout = (_Float16*)alloc((size_t)4 * 8 * 512 * 2);
  _Float16* p_ew1    = (_Float16*)alloc((size_t)NLAYER * 9 * 8 * 512 * 2);
  _Float16* p_ew2    = (_Float16*)alloc((size_t)NLAYER * 4 * 8 * 512 * 2);
  _Float16* p_cw1    = (_Float16*)alloc((size_t)NLAYER * 4 * 8 * 512 * 2);
  _Float16* p_nw1    = (_Float16*)alloc((size_t)NLAYER * 8 * 8 * 512 * 2);
  _Float16* p_nw2    = (_Float16*)alloc((size_t)NLAYER * 4 * 8 * 512 * 2);
  _Float16* p_rw1    = (_Float16*)alloc((size_t)2 * 4 * 8 * 512 * 2);
  _Float16* p_rw2    = (_Float16*)alloc((size_t)2 * 4 * 8 * 512 * 2);
  _Float16* p_hw     = (_Float16*)alloc((size_t)4 * 63 * 512 * 2);

  auto pack = [&](const float* W, _Float16* P, int K, int Nf, int KT, int NT) {
    int total = KT * NT * 512;
    pack_w_kernel<<<(total + 255) / 256, 256, 0, stream>>>(W, P, K, Nf, KT, NT);
  };

  // init state
  fill_kernel<<<(N_NODES * HDIM + 255) / 256, 256, 0, stream>>>(agg, 0.f, N_NODES * HDIM);
  fill_kernel<<<(N_NODES * 3 + 255) / 256, 256, 0, stream>>>(posacc, 0.f, N_NODES * 3);
  fill_kernel<<<(N_NODES + 255) / 256, 256, 0, stream>>>(cnt, 0.f, N_NODES);
  fill_kernel<<<(N_BATCH * HDIM + 255) / 256, 256, 0, stream>>>(g, -INFINITY, N_BATCH * HDIM);
  (void)hipMemcpyAsync(posw, pos_in, (size_t)N_NODES * 3 * 4, hipMemcpyDeviceToDevice, stream);
  count_kernel<<<(N_EDGES + 255) / 256, 256, 0, stream>>>(row, cnt, N_EDGES);

  // pack weights into WMMA B-fragment order (f16)
  pack(emb_in_w, p_embin, XDIM, HDIM, 2, 8);
  pack(emb_out_w, p_embout, HDIM, HDIM, 4, 8);
  for (int l = 0; l < NLAYER; ++l) {
    pack(edge_w1 + (size_t)l * 273 * HDIM, p_ew1 + (size_t)l * 9 * 8 * 512, 273, HDIM, 9, 8);
    pack(edge_w2 + (size_t)l * HDIM * HDIM, p_ew2 + (size_t)l * 4 * 8 * 512, HDIM, HDIM, 4, 8);
    pack(coord_w1 + (size_t)l * HDIM * HDIM, p_cw1 + (size_t)l * 4 * 8 * 512, HDIM, HDIM, 4, 8);
    pack(node_w1 + (size_t)l * 256 * HDIM, p_nw1 + (size_t)l * 8 * 8 * 512, 256, HDIM, 8, 8);
    pack(node_w2 + (size_t)l * HDIM * HDIM, p_nw2 + (size_t)l * 4 * 8 * 512, HDIM, HDIM, 4, 8);
  }
  for (int r = 0; r < 2; ++r) {
    pack(res_w1 + (size_t)r * HDIM * HDIM, p_rw1 + (size_t)r * 4 * 8 * 512, HDIM, HDIM, 4, 8);
    pack(res_w2 + (size_t)r * HDIM * HDIM, p_rw2 + (size_t)r * 4 * 8 * 512, HDIM, HDIM, 4, 8);
  }
  pack(head_w, p_hw, HDIM, PROPN, 4, 63);

  // dynamic LDS sizes (bytes)
  const size_t sh_embin  = (size_t)(16 * 512 + 8 * 16 * 64) * 2;
  const size_t sh_edge   = (size_t)(72 * 512 + 32 * 512 + 32 * 512 + 8 * 4608 + 8 * 2048) * 2;
  const size_t sh_node   = (size_t)(64 * 512 + 32 * 512 + 8 * 4096 + 8 * 2048) * 2;
  const size_t sh_embout = (size_t)(32 * 512 + 8 * 2048) * 2;
  const size_t sh_head   = (size_t)(2 * 32 * 512 + 2 * 32 * 512 + 8 * 2048 + 8 * 2048) * 2;

  emb_in_kernel<<<160, 256, sh_embin, stream>>>(x, p_embin, emb_in_b, h, N_NODES);

  for (int l = 0; l < NLAYER; ++l) {
    edge_kernel<<<512, 256, sh_edge, stream>>>(
        h, posw, edge_attr, row, col,
        p_ew1 + (size_t)l * 9 * 8 * 512, edge_b1 + (size_t)l * HDIM,
        p_ew2 + (size_t)l * 4 * 8 * 512, edge_b2 + (size_t)l * HDIM,
        p_cw1 + (size_t)l * 4 * 8 * 512, coord_b1 + (size_t)l * HDIM,
        coord_w2 + (size_t)l * HDIM,
        agg, posacc, N_EDGES);
    pos_update_kernel<<<(N_NODES * 3 + 255) / 256, 256, 0, stream>>>(posw, posacc, cnt, N_NODES);
    node_kernel<<<160, 256, sh_node, stream>>>(
        h, agg,
        p_nw1 + (size_t)l * 8 * 8 * 512, node_b1 + (size_t)l * HDIM,
        p_nw2 + (size_t)l * 4 * 8 * 512, node_b2 + (size_t)l * HDIM,
        N_NODES);
  }

  emb_out_pool_kernel<<<160, 256, sh_embout, stream>>>(h, p_embout, emb_out_b, batch, g, N_NODES);
  head_kernel<<<8, 256, sh_head, stream>>>(g, p_rw1, res_b1, p_rw2, res_b2, p_hw, head_b, out);
}
